// LSTMDecoder_75359496175946
// MI455X (gfx1250) — matrix operand order repaired and checked by reading, not compiled, not run on hardware
//
#include <hip/hip_runtime.h>
#include <hip/hip_bf16.h>

// ---------------------------------------------------------------------------
// Persistent LSTM decoder for MI455X (gfx1250, wave32, WMMA).
// B=256, S=1024, H=256.
//
// One persistent kernel runs all 1024 recurrence steps (software grid barrier
// across 64 resident workgroups) instead of 1024 launches. Per step:
//   gates(256x1024) = [tok | h] (256x512 bf16) @ [W_ih ; W_hh]^T (1024x512 bf16)
// via v_wmma_f32_16x16x32_bf16.  B-fragments come from LDS (staged once,
// 64 KB/WG); cell state c lives in registers for the whole sequence.
// ---------------------------------------------------------------------------

typedef __attribute__((ext_vector_type(16))) __bf16 v16bf;
typedef __attribute__((ext_vector_type(8)))  __bf16 v8bf;
typedef __attribute__((ext_vector_type(8)))  float  v8f;
typedef __attribute__((ext_vector_type(4)))  float  f32x4;

#define LSTM_B 256
#define LSTM_S 1024
#define LSTM_H 256
#define LSTM_K 512            // concat K = 2*H
#define NBLK   64             // 16 N-blocks x 4 M-blocks, all co-resident

__device__ __forceinline__ float sigm_f(float x) {
    return 1.0f / (1.0f + __expf(-x));
}
__device__ __forceinline__ float tanh_f(float x) {
    float e = __expf(2.0f * x);
    return (e - 1.0f) / (e + 1.0f);
}

// A fragment (16x32 bf16, CDNA5 wave32 layout) from an fp32 row-major matrix.
// lanes 0-15: M=lane, K = k0..k0+7 (v0..3) and k0+16..k0+23 (v4..7)
// lanes16-31: M=lane-16, K = k0+8..k0+15 and k0+24..k0+31
__device__ __forceinline__ v16bf a_frag_f32(const float* base, long stride,
                                            int Mbase, int k0, int lane) {
    const float* p = base + (long)(Mbase + (lane & 15)) * stride
                          + k0 + ((lane >> 4) << 3);
    f32x4 x0 = *(const f32x4*)(p);
    f32x4 x1 = *(const f32x4*)(p + 4);
    f32x4 x2 = *(const f32x4*)(p + 16);
    f32x4 x3 = *(const f32x4*)(p + 20);
    v16bf f;
    f[0]  = (__bf16)x0[0]; f[1]  = (__bf16)x0[1]; f[2]  = (__bf16)x0[2]; f[3]  = (__bf16)x0[3];
    f[4]  = (__bf16)x1[0]; f[5]  = (__bf16)x1[1]; f[6]  = (__bf16)x1[2]; f[7]  = (__bf16)x1[3];
    f[8]  = (__bf16)x2[0]; f[9]  = (__bf16)x2[1]; f[10] = (__bf16)x2[2]; f[11] = (__bf16)x2[3];
    f[12] = (__bf16)x3[0]; f[13] = (__bf16)x3[1]; f[14] = (__bf16)x3[2]; f[15] = (__bf16)x3[3];
    return f;
}

// A fragment from a bf16 row-major matrix (two b128 loads per lane).
__device__ __forceinline__ v16bf a_frag_bf16(const __bf16* base, int stride,
                                             int Mbase, int k0, int lane) {
    const __bf16* p = base + (Mbase + (lane & 15)) * stride
                           + k0 + ((lane >> 4) << 3);
    v8bf lo = *(const v8bf*)(p);
    v8bf hi = *(const v8bf*)(p + 16);
    return __builtin_shufflevector(lo, hi, 0, 1, 2, 3, 4, 5, 6, 7,
                                           8, 9, 10, 11, 12, 13, 14, 15);
}

// Persistent kernel.  Grid MUST be (16, 4) x 128 threads: 64 workgroups,
// all simultaneously resident (MI455X has far more WGPs), synchronized by a
// monotonic atomic counter per step.
__global__ __launch_bounds__(128) void
lstm_persistent_kernel(const float* __restrict__ x,      // (B,S,H) f32
                       const float* __restrict__ h_n,    // (B,H) f32
                       const float* __restrict__ c_n,    // (B,H) f32
                       const __bf16* __restrict__ Wcat,  // (4H, 2H) bf16
                       const float* __restrict__ bias,   // (4H) = b_ih+b_hh
                       __bf16* __restrict__ hbuf0,       // (B,H) bf16 ping
                       __bf16* __restrict__ hbuf1,       // (B,H) bf16 pong
                       float* __restrict__ out,          // (B,S,H) f32
                       unsigned* __restrict__ bar) {     // grid barrier ctr
    // 64 weight rows (4 gates x 16 columns of this N-block), full K: 64 KB.
    __shared__ __bf16 Wlds[64][LSTM_K];

    const int lane  = threadIdx.x & 31;
    const int wave  = threadIdx.x >> 5;
    const int Mbase = blockIdx.y * 64 + wave * 16;
    const int Nbase = blockIdx.x * 16;
    const int ncol  = Nbase + (lane & 15);

    // ---- Stage this block's weight rows into LDS (once) -------------------
    // 64 rows x 512 bf16 = 4096 chunks of 8 bf16 (16B); 32 chunks per thread.
    for (int c = threadIdx.x; c < (64 * LSTM_K) / 8; c += 128) {
        const int r   = c >> 6;             // LDS row 0..63
        const int off = (c & 63) << 3;      // K offset, 8-elem chunks
        const int g   = r >> 4;             // gate 0..3
        const int col = Nbase + (r & 15);   // gate column
        *(v8bf*)&Wlds[r][off] =
            *(const v8bf*)(Wcat + (size_t)(g * LSTM_H + col) * LSTM_K + off);
    }
    __syncthreads();

    // ---- Per-thread persistent state --------------------------------------
    const float bi = bias[0 * LSTM_H + ncol];
    const float bf = bias[1 * LSTM_H + ncol];
    const float bg = bias[2 * LSTM_H + ncol];
    const float bo = bias[3 * LSTM_H + ncol];

    // Cell state stays in VGPRs for the entire sequence (never re-read from
    // memory; the reference does not output c).
    float creg[8];
#pragma unroll
    for (int r = 0; r < 8; ++r) {
        const int row = Mbase + r + ((lane >> 4) << 3);
        creg[r] = c_n[row * LSTM_H + ncol];
    }

    // ---- Sequence loop -----------------------------------------------------
    for (int s = 0; s < LSTM_S; ++s) {
        const float* tok      = (s == 0) ? h_n : (x + (long)(s - 1) * LSTM_H);
        const long   tokStride = (s == 0) ? (long)LSTM_H
                                          : (long)LSTM_S * LSTM_H;
        const __bf16* hprev = (s & 1) ? hbuf1 : hbuf0;
        __bf16*       hnext = (s & 1) ? hbuf0 : hbuf1;

        v8f acc[4];
#pragma unroll
        for (int g = 0; g < 4; ++g)
            acc[g] = (v8f){0.f, 0.f, 0.f, 0.f, 0.f, 0.f, 0.f, 0.f};

        // K chunks 0..7: token (fp32 -> bf16 in-register)
#pragma unroll
        for (int kc = 0; kc < 8; ++kc) {
            const int k0 = kc * 32;
            v16bf a = a_frag_f32(tok, tokStride, Mbase, k0, lane);
            const int kk = k0 + ((lane >> 4) << 3);
#pragma unroll
            for (int g = 0; g < 4; ++g) {
                const int r = g * 16 + (lane & 15);
                v8bf lo = *(const v8bf*)&Wlds[r][kk];
                v8bf hi = *(const v8bf*)&Wlds[r][kk + 16];
                v16bf b = __builtin_shufflevector(lo, hi, 0, 1, 2, 3, 4, 5, 6, 7,
                                                  8, 9, 10, 11, 12, 13, 14, 15);
                acc[g] = __builtin_amdgcn_wmma_f32_16x16x32_bf16(
                    false, a, false, b, (short)0, acc[g], false, false);
            }
        }
        // K chunks 8..15: previous hidden state (bf16 ping-pong buffer)
#pragma unroll
        for (int kc = 8; kc < 16; ++kc) {
            const int k0 = kc * 32;
            v16bf a = a_frag_bf16(hprev, LSTM_H, Mbase, k0 - LSTM_H, lane);
            const int kk = k0 + ((lane >> 4) << 3);
#pragma unroll
            for (int g = 0; g < 4; ++g) {
                const int r = g * 16 + (lane & 15);
                v8bf lo = *(const v8bf*)&Wlds[r][kk];
                v8bf hi = *(const v8bf*)&Wlds[r][kk + 16];
                v16bf b = __builtin_shufflevector(lo, hi, 0, 1, 2, 3, 4, 5, 6, 7,
                                                  8, 9, 10, 11, 12, 13, 14, 15);
                acc[g] = __builtin_amdgcn_wmma_f32_16x16x32_bf16(
                    false, a, false, b, (short)0, acc[g], false, false);
            }
        }

        // Epilogue: gate activations, register-resident cell update, h writes.
        float* outS = out + (long)s * LSTM_H;
#pragma unroll
        for (int r = 0; r < 8; ++r) {
            const int row = Mbase + r + ((lane >> 4) << 3);
            const float gi = acc[0][r] + bi;
            const float gf = acc[1][r] + bf;
            const float gg = acc[2][r] + bg;
            const float go = acc[3][r] + bo;
            const float cn = sigm_f(gf) * creg[r] + sigm_f(gi) * tanh_f(gg);
            const float hv = sigm_f(go) * tanh_f(cn);
            creg[r] = cn;
            hnext[row * LSTM_H + ncol] = (__bf16)hv;
            outS[(long)row * (LSTM_S * LSTM_H) + ncol] = hv;
        }

        // ---- Grid-wide software barrier (monotonic counter, no reset) -----
        __threadfence();            // release our h stores (agent scope)
        __syncthreads();
        if (threadIdx.x == 0) {
            __hip_atomic_fetch_add(bar, 1u, __ATOMIC_ACQ_REL,
                                   __HIP_MEMORY_SCOPE_AGENT);
            const unsigned target = (unsigned)(s + 1) * (unsigned)NBLK;
            while (__hip_atomic_load(bar, __ATOMIC_ACQUIRE,
                                     __HIP_MEMORY_SCOPE_AGENT) < target)
                __builtin_amdgcn_s_sleep(1);
        }
        __syncthreads();
        __threadfence();            // acquire: see other WGs' h stores
    }
}

// Prep: concat+convert weights to bf16, fuse biases, init h0 (bf16),
// zero the grid-barrier counter.  Runs every call (deterministic).
__global__ void lstm_prep_kernel(const float* __restrict__ W_ih,
                                 const float* __restrict__ W_hh,
                                 const float* __restrict__ b_ih,
                                 const float* __restrict__ b_hh,
                                 const float* __restrict__ h_n,
                                 __bf16* __restrict__ Wcat,
                                 float* __restrict__ bias,
                                 __bf16* __restrict__ h0buf,
                                 unsigned* __restrict__ bar) {
    const int idx    = blockIdx.x * blockDim.x + threadIdx.x;
    const int stride = gridDim.x * blockDim.x;
    if (idx == 0) *bar = 0u;
    const int wtot = 4 * LSTM_H * LSTM_K;  // 1024 * 512
    for (int i = idx; i < wtot; i += stride) {
        const int g = i >> 9;          // gate row 0..1023
        const int k = i & 511;         // concat-K
        const float v = (k < LSTM_H) ? W_ih[g * LSTM_H + k]
                                     : W_hh[g * LSTM_H + (k - LSTM_H)];
        Wcat[i] = (__bf16)v;
    }
    for (int i = idx; i < 4 * LSTM_H; i += stride)
        bias[i] = b_ih[i] + b_hh[i];
    for (int i = idx; i < LSTM_B * LSTM_H; i += stride)
        h0buf[i] = (__bf16)h_n[i];
}

extern "C" void kernel_launch(void* const* d_in, const int* in_sizes, int n_in,
                              void* d_out, int out_size, void* d_ws, size_t ws_size,
                              hipStream_t stream) {
    const float* x    = (const float*)d_in[0];  // (B,S,H)
    const float* h_n  = (const float*)d_in[1];  // (1,B,H)
    const float* c_n  = (const float*)d_in[2];  // (1,B,H)
    const float* W_ih = (const float*)d_in[3];  // (4H,H)
    const float* W_hh = (const float*)d_in[4];  // (4H,H)
    const float* b_ih = (const float*)d_in[5];  // (4H)
    const float* b_hh = (const float*)d_in[6];  // (4H)
    float* out = (float*)d_out;                 // (B,S,H)

    // Workspace layout (256B-aligned, ~1.3 MB total)
    char* w = (char*)d_ws;
    size_t off = 0;
    __bf16* Wcat = (__bf16*)(w + off); off += (size_t)4 * LSTM_H * LSTM_K * 2;  // 1 MiB
    float*  bias = (float*) (w + off); off += (size_t)4 * LSTM_H * 4;
    off = (off + 255) & ~(size_t)255;
    __bf16* hbuf0 = (__bf16*)(w + off); off += (size_t)LSTM_B * LSTM_H * 2;
    off = (off + 255) & ~(size_t)255;
    __bf16* hbuf1 = (__bf16*)(w + off); off += (size_t)LSTM_B * LSTM_H * 2;
    off = (off + 255) & ~(size_t)255;
    unsigned* bar = (unsigned*)(w + off); off += 256;

    lstm_prep_kernel<<<1024, 256, 0, stream>>>(W_ih, W_hh, b_ih, b_hh, h_n,
                                               Wcat, bias, hbuf0, bar);

    lstm_persistent_kernel<<<dim3(LSTM_H / 16, LSTM_B / 64), 128, 0, stream>>>(
        x, h_n, c_n, Wcat, bias, hbuf0, hbuf1, out, bar);
}
